// SparseGraphAttention_80169859547197
// MI455X (gfx1250) — matrix operand order repaired and checked by reading, not compiled, not run on hardware
//
#include <hip/hip_runtime.h>
#include <hip/hip_bf16.h>
#include <stddef.h>
#include <stdint.h>

// ---------- geometry -------------------------------------------------------
#define BB 4
#define LL 512
#define DNODE 128
#define DPAIR 64
#define NHEAD 4
#define DH 32
#define KKNN 12
#define NSS 4
#define EE 16
#define ROWS (BB * LL)   // 2048

typedef _Float16 v16h __attribute__((ext_vector_type(16)));
typedef _Float16 v8h  __attribute__((ext_vector_type(8)));
typedef float    v8f  __attribute__((ext_vector_type(8)));

// ---------- kernel 1: kNN + top-SS graph build -----------------------------
__global__ void build_graph_kernel(const float* __restrict__ coords,
                                   const float* __restrict__ css,
                                   int* __restrict__ eidx) {
    int t = blockIdx.x * blockDim.x + threadIdx.x;
    if (t >= ROWS) return;
    int b = t >> 9, i = t & (LL - 1);
    const float* cb = coords + (size_t)b * LL * 3;
    float xi = cb[i * 3 + 0], yi = cb[i * 3 + 1], zi = cb[i * 3 + 2];

    float bd[KKNN]; int bi[KKNN];
#pragma unroll
    for (int k = 0; k < KKNN; ++k) { bd[k] = 3.0e38f; bi[k] = 0; }
    for (int j = 0; j < LL; ++j) {
        if (j == i) continue;                      // self masked (ref adds 1e9)
        float dx = cb[j * 3 + 0] - xi;
        float dy = cb[j * 3 + 1] - yi;
        float dz = cb[j * 3 + 2] - zi;
        float d = dx * dx + dy * dy + dz * dz;     // same ordering as sqrt(d)
        if (d < bd[KKNN - 1]) {
            int p = KKNN - 1;
            while (p > 0 && bd[p - 1] > d) { bd[p] = bd[p - 1]; bi[p] = bi[p - 1]; --p; }
            bd[p] = d; bi[p] = j;
        }
    }
    float sd[NSS]; int si[NSS];
#pragma unroll
    for (int k = 0; k < NSS; ++k) { sd[k] = -3.0e38f; si[k] = 0; }
    const float* cr = css + ((size_t)b * LL + i) * LL;
    for (int j = 0; j < LL; ++j) {
        float s = cr[j];
        if (s > sd[NSS - 1]) {
            int p = NSS - 1;
            while (p > 0 && sd[p - 1] < s) { sd[p] = sd[p - 1]; si[p] = si[p - 1]; --p; }
            sd[p] = s; si[p] = j;
        }
    }
    int* o = eidx + (size_t)t * EE;
#pragma unroll
    for (int k = 0; k < KKNN; ++k) o[k] = bi[k];
#pragma unroll
    for (int k = 0; k < NSS; ++k) o[KKNN + k] = si[k];
}

// ---------- kernel 2: layernorm -> f16 -------------------------------------
__global__ void ln_to_f16_kernel(const float* __restrict__ x,
                                 const float* __restrict__ gamma,
                                 const float* __restrict__ beta,
                                 _Float16* __restrict__ out) {
    __shared__ float red[DNODE];
    int row = blockIdx.x, d = threadIdx.x;
    float v = x[(size_t)row * DNODE + d];
    red[d] = v;
    __syncthreads();
    for (int s = DNODE / 2; s > 0; s >>= 1) {
        if (d < s) red[d] += red[d + s];
        __syncthreads();
    }
    float mu = red[0] * (1.0f / DNODE);
    __syncthreads();
    float dv = v - mu;
    red[d] = dv * dv;
    __syncthreads();
    for (int s = DNODE / 2; s > 0; s >>= 1) {
        if (d < s) red[d] += red[d + s];
        __syncthreads();
    }
    float var = red[0] * (1.0f / DNODE);
    float y = dv * rsqrtf(var + 1e-5f) * gamma[d] + beta[d];
    out[(size_t)row * DNODE + d] = (_Float16)y;
}

// ---------- kernel 3: f32 -> f16 weight convert ----------------------------
__global__ void f32_to_f16_kernel(const float* __restrict__ src,
                                  _Float16* __restrict__ dst, int n) {
    int i = blockIdx.x * blockDim.x + threadIdx.x;
    if (i < n) dst[i] = (_Float16)src[i];
}

// ---------- kernel 4: WMMA GEMM  O = A(f16, MxK) @ W(f16, NxK)^T -----------
// Block = 256 threads = 8 waves; all waves share one 16-row A tile (staged in
// LDS once via CDNA5 async global->LDS), each wave owns a 16-col slice of N.
// K=128 -> 4x v_wmma_f32_16x16x32_f16 per wave.
template <bool EPI>
__global__ void gemm_wmma_kernel(const _Float16* __restrict__ A,
                                 const _Float16* __restrict__ W0,
                                 const _Float16* __restrict__ W1,
                                 const _Float16* __restrict__ W2,
                                 float* __restrict__ O0,
                                 float* __restrict__ O1,
                                 float* __restrict__ O2,
                                 const float* __restrict__ residual,
                                 const float* __restrict__ bias) {
    __shared__ _Float16 sA[16 * DNODE];        // 4 KB A tile
    const int z = blockIdx.z;
    const _Float16* __restrict__ W = (z == 0) ? W0 : (z == 1) ? W1 : W2;
    float* __restrict__ O = (z == 0) ? O0 : (z == 1) ? O1 : O2;

    const int tid   = threadIdx.x;             // 0..255
    const int lane  = tid & 31;
    const int wave  = tid >> 5;
    const int tileM = blockIdx.x * 16;
    const int tileN = wave * 16;
    const int Kd = DNODE;

    // ---- async-stage the shared A tile: 256 threads x 16B = 4096B ----
    {
        uint32_t lds_dst = (uint32_t)(uintptr_t)(&sA[tid * 8]);
        uint64_t gsrc = (uint64_t)(uintptr_t)(A + (size_t)tileM * Kd + tid * 8);
        asm volatile("global_load_async_to_lds_b128 %0, %1, off"
                     :: "v"(lds_dst), "v"(gsrc) : "memory");
        asm volatile("s_wait_asynccnt 0" ::: "memory");
    }
    __syncthreads();

    const int n     = tileN + (lane & 15);
    const int hiA   = (lane >= 16) ? 16 : 0;   // A layout: hi half-wave holds K+16
    const int kbB   = (lane >= 16) ? 16 : 0;   // B layout: hi half-wave holds K+16

    const _Float16* arow = &sA[(lane & 15) * Kd];
    const _Float16* wrow = W + (size_t)n * Kd;

    v8f c = {};
#pragma unroll
    for (int kk = 0; kk < Kd; kk += 32) {
        if (kk + 32 < Kd) __builtin_prefetch(wrow + kk + 32 + kbB, 0, 1);
        v8h a0 = *(const v8h*)(arow + kk + hiA);          // ds_load_b128
        v8h a1 = *(const v8h*)(arow + kk + hiA + 8);
        v8h b0 = *(const v8h*)(wrow + kk + kbB);          // global_load_b128
        v8h b1 = *(const v8h*)(wrow + kk + kbB + 8);
        v16h av = __builtin_shufflevector(a0, a1, 0, 1, 2, 3, 4, 5, 6, 7,
                                          8, 9, 10, 11, 12, 13, 14, 15);
        v16h bv = __builtin_shufflevector(b0, b1, 0, 1, 2, 3, 4, 5, 6, 7,
                                          8, 9, 10, 11, 12, 13, 14, 15);
        c = __builtin_amdgcn_wmma_f32_16x16x32_f16(false, av, false, bv,
                                                   (short)0, c, false, false);
    }

    const int rowBase = tileM + ((lane >= 16) ? 8 : 0);
    const int col     = tileN + (lane & 15);
    if constexpr (EPI) {
        float bcol = bias[col];
#pragma unroll
        for (int r = 0; r < 8; ++r) {
            int row = rowBase + r;
            O[(size_t)row * DNODE + col] =
                c[r] + bcol + residual[(size_t)row * DNODE + col];
        }
    } else {
#pragma unroll
        for (int r = 0; r < 8; ++r) {
            int row = rowBase + r;
            O[(size_t)row * DNODE + col] = c[r];
        }
    }
}

// ---------- kernel 5: sparse attention (E=16 neighbors) --------------------
// One block (128 threads) per (b,l).
__global__ void attn_kernel(const float* __restrict__ Q,
                            const float* __restrict__ K,
                            const float* __restrict__ V,
                            const int* __restrict__ eidx,
                            const float* __restrict__ pair,
                            const float* __restrict__ W_eb,
                            _Float16* __restrict__ ctx) {
    __shared__ float s_sc[NHEAD][EE];
    __shared__ int   s_e[EE];
    __shared__ float s_q[DNODE];

    const int bl = blockIdx.x;                 // 0 .. ROWS-1
    const int b = bl >> 9, l = bl & (LL - 1);
    const int tid = threadIdx.x;

    if (tid < EE) s_e[tid] = eidx[(size_t)bl * EE + tid];
    s_q[tid] = Q[(size_t)bl * DNODE + tid];
    __syncthreads();

    if (tid < NHEAD * EE) {                    // 64 threads: one (h,e) each
        int h = tid >> 4, e = tid & (EE - 1);
        int j = s_e[e];
        const float* kr = K + ((size_t)b * LL + j) * DNODE + h * DH;
        float acc = 0.0f;
#pragma unroll
        for (int d = 0; d < DH; ++d) acc += s_q[h * DH + d] * kr[d];
        acc *= 0.17677669529663687f;           // DH^-0.5
        const float* pr = pair + (((size_t)b * LL + l) * LL + j) * DPAIR;
        const float* wr = W_eb + h * DPAIR;
        float bia = 0.0f;
#pragma unroll 8
        for (int d = 0; d < DPAIR; ++d) bia += pr[d] * wr[d];
        s_sc[h][e] = acc + bia;
    }
    __syncthreads();

    const int h = tid >> 5, d = tid & (DH - 1);
    float mx = -3.0e38f;
#pragma unroll
    for (int e = 0; e < EE; ++e) mx = fmaxf(mx, s_sc[h][e]);
    float den = 0.0f, acc = 0.0f;
#pragma unroll
    for (int e = 0; e < EE; ++e) {
        float w = __expf(s_sc[h][e] - mx);
        den += w;
        int j = s_e[e];
        acc += w * V[((size_t)b * LL + j) * DNODE + h * DH + d];
    }
    ctx[(size_t)bl * DNODE + tid] = (_Float16)(acc / den);
}

// ---------- launcher -------------------------------------------------------
extern "C" void kernel_launch(void* const* d_in, const int* in_sizes, int n_in,
                              void* d_out, int out_size, void* d_ws, size_t ws_size,
                              hipStream_t stream) {
    const float* single = (const float*)d_in[0];
    const float* pair   = (const float*)d_in[1];
    const float* coords = (const float*)d_in[2];
    const float* css    = (const float*)d_in[3];
    const float* gamma  = (const float*)d_in[4];
    const float* beta   = (const float*)d_in[5];
    const float* Wq     = (const float*)d_in[6];
    const float* Wk     = (const float*)d_in[7];
    const float* Wv     = (const float*)d_in[8];
    const float* W_eb   = (const float*)d_in[9];
    const float* Wo     = (const float*)d_in[10];
    const float* bo     = (const float*)d_in[11];
    float* out = (float*)d_out;

    // workspace carve-up (256B aligned blocks)
    char* ws = (char*)d_ws;
    int*       eidx  = (int*)(ws);                       // 2048*16*4      = 128 KiB
    _Float16*  sn16  = (_Float16*)(ws + (131072));       // 2048*128*2     = 512 KiB
    _Float16*  Wq16  = (_Float16*)(ws + (655360));       // 128*128*2 each
    _Float16*  Wk16  = (_Float16*)(ws + (688128));
    _Float16*  Wv16  = (_Float16*)(ws + (720896));
    _Float16*  Wo16  = (_Float16*)(ws + (753664));
    float*     Qb    = (float*)(ws + (786432));          // 2048*128*4 each
    float*     Kb    = (float*)(ws + (1835008));
    float*     Vb    = (float*)(ws + (2883584));
    _Float16*  ctx16 = (_Float16*)(ws + (3932160));      // 2048*128*2

    // 1. graph build
    build_graph_kernel<<<dim3((ROWS + 255) / 256), dim3(256), 0, stream>>>(coords, css, eidx);
    // 2. layernorm -> f16
    ln_to_f16_kernel<<<dim3(ROWS), dim3(DNODE), 0, stream>>>(single, gamma, beta, sn16);
    // 3. weight converts
    f32_to_f16_kernel<<<dim3(64), dim3(256), 0, stream>>>(Wq, Wq16, DNODE * DNODE);
    f32_to_f16_kernel<<<dim3(64), dim3(256), 0, stream>>>(Wk, Wk16, DNODE * DNODE);
    f32_to_f16_kernel<<<dim3(64), dim3(256), 0, stream>>>(Wv, Wv16, DNODE * DNODE);
    f32_to_f16_kernel<<<dim3(64), dim3(256), 0, stream>>>(Wo, Wo16, DNODE * DNODE);
    // 4. Q/K/V projections (WMMA), grid.z selects the weight
    gemm_wmma_kernel<false><<<dim3(ROWS / 16, 1, 3), dim3(256), 0, stream>>>(
        sn16, Wq16, Wk16, Wv16, Qb, Kb, Vb, nullptr, nullptr);
    // 5. sparse attention
    attn_kernel<<<dim3(ROWS), dim3(128), 0, stream>>>(Qb, Kb, Vb, eidx, pair, W_eb, ctx16);
    // 6. output projection + residual + bias (WMMA)
    gemm_wmma_kernel<true><<<dim3(ROWS / 16, 1, 1), dim3(256), 0, stream>>>(
        ctx16, Wo16, Wo16, Wo16, out, out, out, single, bo);
}
